// basic_gnn_25082609009166
// MI455X (gfx1250) — matrix operand order repaired and checked by reading, not compiled, not run on hardware
//
#include <hip/hip_runtime.h>
#include <math.h>

// ---------------------------------------------------------------------------
// CDNA5 (gfx1250) GCN forward pass.
//   - Dense linear layers: v_wmma_f32_16x16x32_f16 (f16 A/B, f32 accumulate)
//   - Aggregation: wave-per-edge gather + global_atomic_add_f32 scatter (L2)
//   - Feature buffers padded to 64-row multiples -> unconditional WMMA stores
// ---------------------------------------------------------------------------

typedef __attribute__((ext_vector_type(16))) _Float16 v16h;
typedef __attribute__((ext_vector_type(8)))  float    v8f;

#define FEAT 128  // H1 == H2 == 128

// Hardware fp32 atomic add (global_atomic_add_f32 at L2; no CAS loop).
__device__ __forceinline__ void atomic_add_f32(float* p, float v) {
    unsafeAtomicAdd(p, v);
}

// ---------------------------- small utility kernels ------------------------

__global__ void fill_f32_kernel(float* p, float v, int n) {
    int i = blockIdx.x * blockDim.x + threadIdx.x;
    for (; i < n; i += gridDim.x * blockDim.x) p[i] = v;
}

// deg[col[e]] += ew[e]   (deg pre-filled with 1.0 for the self loop)
__global__ void deg_edges_kernel(float* deg, const int* col, const float* ew, int E) {
    int e = blockIdx.x * blockDim.x + threadIdx.x;
    for (; e < E; e += gridDim.x * blockDim.x) atomic_add_f32(&deg[col[e]], ew[e]);
}

// dis = deg > 0 ? rsqrt(deg) : 0   (in place)
__global__ void rsqrt_kernel(float* d, int n) {
    int i = blockIdx.x * blockDim.x + threadIdx.x;
    for (; i < n; i += gridDim.x * blockDim.x) {
        float v = d[i];
        d[i] = (v > 0.0f) ? rsqrtf(fmaxf(v, 1e-30f)) : 0.0f;
    }
}

// norm[e] = dis[row]*ew*dis[col]
__global__ void norm_kernel(float* nrm, const float* dis, const int* row,
                            const int* col, const float* ew, int E) {
    int e = blockIdx.x * blockDim.x + threadIdx.x;
    for (; e < E; e += gridDim.x * blockDim.x)
        nrm[e] = dis[row[e]] * ew[e] * dis[col[e]];
}

// ------------------------------- WMMA GEMM ---------------------------------
// out[np x 128] = act(X[n x K]) @ W[K x 128], staged through LDS as f16.
// Block: 128 threads = 4 waves; each block covers 64 rows; each wave 16 rows.
// `out` must have capacity for gridDim.x*64 rows (padded) -> unconditional
// stores, no exec-mask ladders. Rows past n are zero-padded on load.
template <int K, bool RELU>
__global__ __launch_bounds__(128) void gemm_wmma_kernel(
    const float* __restrict__ X, const float* __restrict__ W,
    float* __restrict__ out, int n) {
    constexpr int F    = FEAT;
    constexpr int ROWS = 64;
    constexpr int NK   = K / 32;

    __shared__ _Float16 Xs[ROWS * K];  // [row][k]
    __shared__ _Float16 Ws[F * K];     // transposed: [ncol][k]

    const int tid = threadIdx.x;
    const int r0  = blockIdx.x * ROWS;

    // Stage X tile (f32 -> f16), optional fused ReLU, zero-pad rows past n.
    for (int idx = tid; idx < ROWS * K; idx += 128) {
        int r = idx / K, k = idx - r * K;
        float v = (r0 + r < n) ? X[(size_t)(r0 + r) * K + k] : 0.0f;
        if (RELU) v = fmaxf(v, 0.0f);
        Xs[idx] = (_Float16)v;
    }
    // Stage W transposed (f32 -> f16).
    for (int idx = tid; idx < K * F; idx += 128) {
        int k = idx / F, c = idx - k * F;
        Ws[c * K + k] = (_Float16)W[idx];
    }
    __syncthreads();

    const int wave = tid >> 5, lane = tid & 31;
    const int m0 = wave * 16;          // wave's row tile inside the block
    const int lm = lane & 15, lg = lane >> 4;

    // Cache all A fragments for this wave's 16 rows (ISA 16-bit A 16x32 layout:
    // lane holds M=lm; element i -> k = c*32 + (i/8)*16 + lg*8 + (i%8)).
    v16h afrag[NK];
#pragma unroll
    for (int c = 0; c < NK; ++c) {
        const _Float16* p = &Xs[(m0 + lm) * K + c * 32 + lg * 8];
        v16h a;
#pragma unroll
        for (int i = 0; i < 8; ++i) { a[i] = p[i]; a[8 + i] = p[16 + i]; }
        afrag[c] = a;
    }

    // 8 column tiles of 16; B layout: lane holds N=lm, element i -> k = c*32 + (i/8)*16 + lg*8 + (i%8).
#pragma unroll
    for (int j = 0; j < F / 16; ++j) {
        v8f acc = {};
#pragma unroll
        for (int c = 0; c < NK; ++c) {
            const _Float16* q = &Ws[(j * 16 + lm) * K + c * 32 + lg * 8];
            v16h b;
#pragma unroll
            for (int i = 0; i < 8; ++i) { b[i] = q[i]; b[8 + i] = q[16 + i]; }
            acc = __builtin_amdgcn_wmma_f32_16x16x32_f16(
                false, afrag[c], false, b, (short)0, acc, false, false);
        }
        // D layout: lane holds col N = j*16+lm; VGPR r -> row M = m0 + lg*8 + r.
        // Unconditional: out is padded to a 64-row multiple.
        float* o = out + (size_t)(r0 + m0 + lg * 8) * F + (j * 16 + lm);
#pragma unroll
        for (int r = 0; r < 8; ++r) o[(size_t)r * F] = acc[r];
    }
}

// --------------------------- aggregation kernels ---------------------------

// agg[i][f] = b[f] + dis[i]^2 * hw[i][f]   (bias + self-loop), float4-wide
__global__ void init_agg_kernel(float4* __restrict__ agg, const float4* __restrict__ bias,
                                const float* __restrict__ dis,
                                const float4* __restrict__ hw, int n) {
    int idx   = blockIdx.x * blockDim.x + threadIdx.x;
    int total = n * (FEAT / 4);
    for (; idx < total; idx += gridDim.x * blockDim.x) {
        int i = idx >> 5, f4 = idx & (FEAT / 4 - 1);
        float sn = dis[i] * dis[i];
        float4 h = hw[idx], b = bias[f4];
        agg[idx] = make_float4(b.x + sn * h.x, b.y + sn * h.y,
                               b.z + sn * h.z, b.w + sn * h.w);
    }
}

// wave per edge: agg[col][:] += norm[e] * hw[row][:]
__global__ void edge_agg_kernel(float* __restrict__ agg, const float* __restrict__ hw,
                                const float* __restrict__ nrm, const int* __restrict__ row,
                                const int* __restrict__ col, int E) {
    int wid  = (blockIdx.x * blockDim.x + threadIdx.x) >> 5;
    int lane = threadIdx.x & 31;
    int nw   = (gridDim.x * blockDim.x) >> 5;
    for (int e = wid; e < E; e += nw) {
        int r = row[e], c = col[e];
        float w = nrm[e];
        const float* src = hw + (size_t)r * FEAT;
        float* dst = agg + (size_t)c * FEAT;
#pragma unroll
        for (int t = 0; t < 4; ++t) {
            int f = lane + 32 * t;
            atomic_add_f32(dst + f, w * src[f]);
        }
    }
}

// ------------------------------ layer-3 path -------------------------------

// s[i] = dot(relu(H[i][0:128]), w3[0:128])   (wave per node, fused ReLU)
__global__ void gemv128_kernel(const float* __restrict__ H, const float* __restrict__ w3,
                               float* __restrict__ s, int n) {
    int wid  = (blockIdx.x * blockDim.x + threadIdx.x) >> 5;
    int lane = threadIdx.x & 31;
    int nw   = (gridDim.x * blockDim.x) >> 5;
    for (int i = wid; i < n; i += nw) {
        float acc = 0.0f;
#pragma unroll
        for (int t = 0; t < 4; ++t) {
            int k = lane + 32 * t;
            acc += fmaxf(H[(size_t)i * FEAT + k], 0.0f) * w3[k];
        }
        for (int off = 16; off > 0; off >>= 1) acc += __shfl_down(acc, off, 32);
        if (lane == 0) s[i] = acc;
    }
}

// p[i] = b3 + dis[i]^2 * s[i]
__global__ void init_agg1_kernel(float* p, const float* b3, const float* dis,
                                 const float* s, int n) {
    int i = blockIdx.x * blockDim.x + threadIdx.x;
    for (; i < n; i += gridDim.x * blockDim.x) {
        float sn = dis[i] * dis[i];
        p[i] = b3[0] + sn * s[i];
    }
}

// p[col[e]] += norm[e] * s[row[e]]
__global__ void edge_agg1_kernel(float* p, const float* s, const float* nrm,
                                 const int* row, const int* col, int E) {
    int e = blockIdx.x * blockDim.x + threadIdx.x;
    for (; e < E; e += gridDim.x * blockDim.x)
        atomic_add_f32(&p[col[e]], nrm[e] * s[row[e]]);
}

// out = LO + (HI-LO)*sigmoid(p);  LO=0, HI=1
__global__ void sigmoid_kernel(float* out, const float* p, int n) {
    int i = blockIdx.x * blockDim.x + threadIdx.x;
    for (; i < n; i += gridDim.x * blockDim.x)
        out[i] = 1.0f / (1.0f + expf(-p[i]));
}

// --------------------------------- launch ----------------------------------

extern "C" void kernel_launch(void* const* d_in, const int* in_sizes, int n_in,
                              void* d_out, int out_size, void* d_ws, size_t ws_size,
                              hipStream_t stream) {
    const float* x   = (const float*)d_in[0];   // [N,64]
    const int*   ei  = (const int*)  d_in[1];   // [2,E]
    const float* ew  = (const float*)d_in[2];   // [E]
    const float* W1  = (const float*)d_in[3];   // [64,128]
    const float* b1  = (const float*)d_in[4];   // [128]
    const float* W2  = (const float*)d_in[5];   // [128,128]
    const float* b2  = (const float*)d_in[6];   // [128]
    const float* W3  = (const float*)d_in[7];   // [128,1]
    const float* b3  = (const float*)d_in[8];   // [1]
    float* out = (float*)d_out;

    const int N  = in_sizes[0] / 64;
    const int E  = in_sizes[2];
    const int Np = (N + 63) & ~63;              // 64-row padded capacity
    const int* row = ei;
    const int* col = ei + E;

    // Workspace partition (floats). hw/ha padded so WMMA stores need no guards.
    float* ws  = (float*)d_ws;
    float* dis = ws;                               // N   (deg then deg^-1/2)
    float* nrm = dis + N;                          // E
    float* hw  = nrm + E;                          // Np*128  (linear output)
    float* ha  = hw + (size_t)Np * FEAT;           // Np*128  (aggregated / h)
    float* s3  = ha + (size_t)Np * FEAT;           // N       (layer-3 linear)
    float* p3  = s3 + N;                           // N       (pre-sigmoid)
    (void)ws_size; (void)n_in; (void)out_size;

    const int T = 256;
    const int gN   = (N + T - 1) / T;
    const int gE   = (E + T - 1) / T;
    const int gNF4 = ((N * (FEAT / 4)) + T - 1) / T;
    const int gGemm  = Np / 64;
    const int gWaveE = 2048;                // grid-stride: 16K waves over E edges
    const int gWaveN = (N + 7) / 8;         // wave per node (8 waves / block)

    // ---- normalization precompute ----
    fill_f32_kernel<<<gN, T, 0, stream>>>(dis, 1.0f, N);               // self loop
    deg_edges_kernel<<<gE, T, 0, stream>>>(dis, col, ew, E);
    rsqrt_kernel<<<gN, T, 0, stream>>>(dis, N);
    norm_kernel<<<gE, T, 0, stream>>>(nrm, dis, row, col, ew, E);

    // ---- layer 1: hw = X@W1 ; ha = agg(hw) + b1 ----
    gemm_wmma_kernel<64, false><<<gGemm, 128, 0, stream>>>(x, W1, hw, N);
    init_agg_kernel<<<gNF4, T, 0, stream>>>((float4*)ha, (const float4*)b1, dis,
                                            (const float4*)hw, N);
    edge_agg_kernel<<<gWaveE, T, 0, stream>>>(ha, hw, nrm, row, col, E);

    // ---- layer 2: hw = relu(ha)@W2 ; ha = agg(hw) + b2 ----  (ReLU fused)
    gemm_wmma_kernel<128, true><<<gGemm, 128, 0, stream>>>(ha, W2, hw, N);
    init_agg_kernel<<<gNF4, T, 0, stream>>>((float4*)ha, (const float4*)b2, dis,
                                            (const float4*)hw, N);
    edge_agg_kernel<<<gWaveE, T, 0, stream>>>(ha, hw, nrm, row, col, E);

    // ---- layer 3: s3 = relu(ha)@W3 ; p3 = agg(s3) + b3 ; sigmoid ----
    gemv128_kernel<<<gWaveN, T, 0, stream>>>(ha, W3, s3, N);
    init_agg1_kernel<<<gN, T, 0, stream>>>(p3, b3, dis, s3, N);
    edge_agg1_kernel<<<gE, T, 0, stream>>>(p3, s3, nrm, row, col, E);
    sigmoid_kernel<<<gN, T, 0, stream>>>(out, p3, N);
}
